// QLinear_69612829933783
// MI455X (gfx1250) — 8-bit matrix operand order repaired and checked by reading, compiled, not run on hardware
//
#include <hip/hip_runtime.h>

typedef int v8i __attribute__((ext_vector_type(8)));

#define B_DIM   4
#define S_DIM   2048
#define IN_DIM  4096
#define OUT_DIM 4096
#define M_DIM   (B_DIM * S_DIM)
#define K_DIM   IN_DIM
#define N_DIM   OUT_DIM

#define BM 128
#define BN 128
#define BK 128
#define KTILES (K_DIM / BK)

#define SCALE_OFF 0
#define XQ_OFF    256
#define WT_OFF    (256 + (size_t)M_DIM * K_DIM)

__device__ __forceinline__ void async_load_b128(unsigned lds_addr,
                                                unsigned gvs_off,
                                                const signed char* sbase) {
    asm volatile("global_load_async_to_lds_b128 %0, %1, %2"
                 :: "v"(lds_addr), "v"(gvs_off), "s"(sbase)
                 : "memory");
}

__device__ __forceinline__ void wait_asynccnt0() {
#if __has_builtin(__builtin_amdgcn_s_wait_asynccnt)
    __builtin_amdgcn_s_wait_asynccnt(0);
#else
    asm volatile("s_wait_asynccnt 0" ::: "memory");
#endif
}

__device__ __forceinline__ unsigned lds_addr_of(const void* p) {
    return (unsigned)(unsigned long long)p;
}

__global__ void zero_scales_kernel(unsigned int* s) {
    if (threadIdx.x < 8) s[threadIdx.x] = 0u;
}

__global__ void absmax_kernel(const float* __restrict__ x,
                              unsigned int* __restrict__ sbits) {
    __shared__ float red[256];
    const int b = blockIdx.y;
    const float4* xb = (const float4*)(x + (size_t)b * S_DIM * IN_DIM);
    const size_t base = (size_t)blockIdx.x * 2048;
    float m = 0.0f;
#pragma unroll
    for (int i = 0; i < 8; ++i) {
        float4 v = xb[base + (size_t)i * 256 + threadIdx.x];
        m = fmaxf(m, fmaxf(fmaxf(fabsf(v.x), fabsf(v.y)),
                           fmaxf(fabsf(v.z), fabsf(v.w))));
    }
    red[threadIdx.x] = m;
    __syncthreads();
    for (int s = 128; s > 0; s >>= 1) {
        if ((int)threadIdx.x < s)
            red[threadIdx.x] = fmaxf(red[threadIdx.x], red[threadIdx.x + s]);
        __syncthreads();
    }
    if (threadIdx.x == 0)
        atomicMax(&sbits[b], __float_as_uint(red[0]));
}

__global__ void quantize_kernel(const float* __restrict__ x,
                                const unsigned int* __restrict__ sbits,
                                signed char* __restrict__ xq) {
    const size_t t    = (size_t)blockIdx.x * blockDim.x + threadIdx.x;
    const size_t base = t * 16;
    const int b = (int)(base / ((size_t)S_DIM * IN_DIM));
    const float inv = 128.0f / __uint_as_float(sbits[b]);
    const float4* xp = (const float4*)(x + base);
    int packed[4];
#pragma unroll
    for (int i = 0; i < 4; ++i) {
        float4 v = xp[i];
        int q0 = max(-128, min(127, (int)rintf(v.x * inv)));
        int q1 = max(-128, min(127, (int)rintf(v.y * inv)));
        int q2 = max(-128, min(127, (int)rintf(v.z * inv)));
        int q3 = max(-128, min(127, (int)rintf(v.w * inv)));
        packed[i] = (q0 & 255) | ((q1 & 255) << 8) | ((q2 & 255) << 16) | ((q3 & 255) << 24);
    }
    *(int4*)(xq + base) = make_int4(packed[0], packed[1], packed[2], packed[3]);
}

__global__ void pack_weight_kernel(const int* __restrict__ wq,
                                   signed char* __restrict__ wt) {
    __shared__ signed char s[32][36];
    const int n0 = blockIdx.x * 32;
    const int k0 = blockIdx.y * 32;
    const int tx = threadIdx.x & 31;
    const int ty = threadIdx.x >> 5;
#pragma unroll
    for (int r = 0; r < 4; ++r) {
        int k = ty + r * 8;
        s[k][tx] = (signed char)wq[(size_t)(k0 + k) * OUT_DIM + n0 + tx];
    }
    __syncthreads();
    const int nl = threadIdx.x >> 3;
    const int kq = (threadIdx.x & 7) * 4;
    unsigned int v = (unsigned int)(unsigned char)s[kq + 0][nl]
                   | ((unsigned int)(unsigned char)s[kq + 1][nl] << 8)
                   | ((unsigned int)(unsigned char)s[kq + 2][nl] << 16)
                   | ((unsigned int)(unsigned char)s[kq + 3][nl] << 24);
    *(unsigned int*)(wt + (size_t)(n0 + nl) * K_DIM + k0 + kq) = v;
}

__device__ __forceinline__ void issue_tile_async(const unsigned* ldsA,
                                                 const unsigned* ldsB,
                                                 const unsigned* goff,
                                                 unsigned koff,
                                                 const signed char* gA,
                                                 const signed char* gB) {
#pragma unroll
    for (int p = 0; p < 4; ++p) {
        async_load_b128(ldsA[p], goff[p] + koff, gA);
        async_load_b128(ldsB[p], goff[p] + koff, gB);
    }
}

__device__ __forceinline__ void compute_tile(const signed char (*__restrict__ sAb)[BK],
                                             const signed char (*__restrict__ sBb)[BK],
                                             v8i (&acc)[2][4],
                                             int wm, int wn, int l16, int khalf) {
#pragma unroll
    for (int k64 = 0; k64 < BK; k64 += 64) {
        v8i afrag[2];
#pragma unroll
        for (int ti = 0; ti < 2; ++ti) {
            const int row = wm * 32 + ti * 16 + l16;
            const signed char* ap = &sAb[row][k64 + khalf * 16];
            int2 c0 = *(const int2*)(ap);
            int2 c1 = *(const int2*)(ap + 8);
            int2 c2 = *(const int2*)(ap + 32);
            int2 c3 = *(const int2*)(ap + 40);
            afrag[ti][0] = c0.x; afrag[ti][1] = c0.y;
            afrag[ti][2] = c1.x; afrag[ti][3] = c1.y;
            afrag[ti][4] = c2.x; afrag[ti][5] = c2.y;
            afrag[ti][6] = c3.x; afrag[ti][7] = c3.y;
        }
        v8i bfrag[4];
#pragma unroll
        for (int tj = 0; tj < 4; ++tj) {
            const int col = wn * 64 + tj * 16 + l16;
            const signed char* bp = &sBb[col][k64 + khalf * 16];
            int4 lo = *(const int4*)(bp);
            int4 hi = *(const int4*)(bp + 32);
            bfrag[tj][0] = lo.x; bfrag[tj][1] = lo.y;
            bfrag[tj][2] = lo.z; bfrag[tj][3] = lo.w;
            bfrag[tj][4] = hi.x; bfrag[tj][5] = hi.y;
            bfrag[tj][6] = hi.z; bfrag[tj][7] = hi.w;
        }
#pragma unroll
        for (int ti = 0; ti < 2; ++ti)
#pragma unroll
            for (int tj = 0; tj < 4; ++tj)
                acc[ti][tj] = __builtin_amdgcn_wmma_i32_16x16x64_iu8(
                     true, afrag[ti],
                     true, bfrag[tj],
                    acc[ti][tj],
                     false,  false);
    }
}

__global__ __launch_bounds__(256) void gemm_iu8_kernel(
        const signed char* __restrict__ xq,
        const signed char* __restrict__ wt,
        const unsigned int* __restrict__ sbits,
        const float* __restrict__ bias,
        float* __restrict__ out) {
    __shared__ __align__(16) signed char sA[2][BM][BK];
    __shared__ __align__(16) signed char sB[2][BN][BK];

    const int tid   = threadIdx.x;
    const int lane  = tid & 31;
    const int wave  = tid >> 5;
    const int wm    = wave & 3;
    const int wn    = wave >> 2;
    const int l16   = lane & 15;
    const int khalf = lane >> 4;

    const int m0 = blockIdx.y * BM;
    const int n0 = blockIdx.x * BN;

    const int srow = tid >> 3;
    const int scol = (tid & 7) * 16;

    const signed char* gA = xq + (size_t)m0 * K_DIM;
    const signed char* gB = wt + (size_t)n0 * K_DIM;

    unsigned goff[4];
    unsigned ldsA[2][4], ldsB[2][4];
#pragma unroll
    for (int p = 0; p < 4; ++p) {
        const int r = p * 32 + srow;
        goff[p]    = (unsigned)(r * K_DIM + scol);
        ldsA[0][p] = lds_addr_of(&sA[0][r][scol]);
        ldsA[1][p] = lds_addr_of(&sA[1][r][scol]);
        ldsB[0][p] = lds_addr_of(&sB[0][r][scol]);
        ldsB[1][p] = lds_addr_of(&sB[1][r][scol]);
    }

    issue_tile_async(ldsA[0], ldsB[0], goff, 0u, gA, gB);
    wait_asynccnt0();
    __syncthreads();

    v8i acc[2][4];
#pragma unroll
    for (int ti = 0; ti < 2; ++ti)
#pragma unroll
        for (int tj = 0; tj < 4; ++tj)
#pragma unroll
            for (int e = 0; e < 8; ++e) acc[ti][tj][e] = 0;

    for (int kt = 0; kt < KTILES; kt += 2) {
        issue_tile_async(ldsA[1], ldsB[1], goff, (unsigned)((kt + 1) * BK), gA, gB);
        if (kt + 2 < KTILES) {
            const int koff2 = (kt + 2) * BK;
            __builtin_prefetch(&gA[(size_t)srow * K_DIM + koff2 + scol], 0, 1);
            __builtin_prefetch(&gB[(size_t)srow * K_DIM + koff2 + scol], 0, 1);
        }
        compute_tile(sA[0], sB[0], acc, wm, wn, l16, khalf);
        wait_asynccnt0();
        __syncthreads();

        if (kt + 2 < KTILES)
            issue_tile_async(ldsA[0], ldsB[0], goff, (unsigned)((kt + 2) * BK), gA, gB);
        if (kt + 3 < KTILES) {
            const int koff3 = (kt + 3) * BK;
            __builtin_prefetch(&gA[(size_t)srow * K_DIM + koff3 + scol], 0, 1);
            __builtin_prefetch(&gB[(size_t)srow * K_DIM + koff3 + scol], 0, 1);
        }
        compute_tile(sA[1], sB[1], acc, wm, wn, l16, khalf);
        if (kt + 2 < KTILES) {
            wait_asynccnt0();
            __syncthreads();
        }
    }

    const int batch = m0 / S_DIM;
    const float fscale = __uint_as_float(sbits[batch]) * (0.01f / 128.0f);
#pragma unroll
    for (int ti = 0; ti < 2; ++ti) {
#pragma unroll
        for (int tj = 0; tj < 4; ++tj) {
            const int colg = n0 + wn * 64 + tj * 16 + l16;
            const float bv = bias[colg];
#pragma unroll
            for (int r = 0; r < 8; ++r) {
                const int rowg = m0 + wm * 32 + ti * 16 + khalf * 8 + r;
                out[(size_t)rowg * N_DIM + colg] =
                    (float)acc[ti][tj][r] * fscale + bv;
            }
        }
    }
}

extern "C" void kernel_launch(void* const* d_in, const int* in_sizes, int n_in,
                              void* d_out, int out_size, void* d_ws, size_t ws_size,
                              hipStream_t stream) {
    const float* x    = (const float*)d_in[0];
    const int*   wq   = (const int*)d_in[1];
    const float* bias = (const float*)d_in[2];
    float*       out  = (float*)d_out;

    char* ws = (char*)d_ws;
    unsigned int* sbits = (unsigned int*)(ws + SCALE_OFF);
    signed char*  xq    = (signed char*)(ws + XQ_OFF);
    signed char*  wt    = (signed char*)(ws + WT_OFF);

    zero_scales_kernel<<<1, 32, 0, stream>>>(sbits);

    absmax_kernel<<<dim3(1024, B_DIM), 256, 0, stream>>>(x, sbits);

    quantize_kernel<<<8192, 256, 0, stream>>>(x, sbits, xq);

    pack_weight_kernel<<<dim3(OUT_DIM / 32, IN_DIM / 32), 256, 0, stream>>>(wq, wt);

    gemm_iu8_kernel<<<dim3(N_DIM / BN, M_DIM / BM), 256, 0, stream>>>(
        xq, wt, sbits, bias, out);
}
